// CommNet_88364657148466
// MI455X (gfx1250) — matrix operand order repaired and checked by reading, not compiled, not run on hardware
//
#include <hip/hip_runtime.h>

// ---------------------------------------------------------------------------
// CDNA5 (gfx1250) CommNet: conv stack as implicit GEMM on
// v_wmma_f32_16x16x32_bf16 (wave32). Halo-padded activations + precomputed
// byte-offset im2col tables (branch/div-free gather in SADDR+voffset32 form);
// each wave computes MT out-channel tiles against one shared B fragment;
// weights double-buffered into LDS via global_load_async_to_lds_b128 +
// s_wait_asynccnt; PReLU + 2x2x2 maxpool fused in an LDS epilogue.
// ---------------------------------------------------------------------------

typedef __attribute__((ext_vector_type(16))) __bf16 v16bf;
typedef __attribute__((ext_vector_type(8)))  float  v8f;
typedef int v4i_t __attribute__((vector_size(16)));   // matches builtin's V4i

union BF16x16 { v16bf v; unsigned short u[16]; };
union Int16   { int4 v[4]; int a[16]; };

#if defined(__AMDGCN__) && __has_builtin(__builtin_amdgcn_global_load_async_to_lds_b128)
#define HAS_ASYNC_LDS 1
#else
#define HAS_ASYNC_LDS 0
#endif

__device__ __forceinline__ unsigned short f32_to_bf16(float f) {
  unsigned int u = __float_as_uint(f);
  unsigned int r = 0x7FFFu + ((u >> 16) & 1u);   // round-to-nearest-even
  return (unsigned short)((u + r) >> 16);
}

// A-matrix (16x32 bf16) per-lane element -> K mapping (ISA 7.12.2):
// lanes 0-15 : V0..V3 = K 0..7,   V4..V7 = K 16..23
// lanes 16-31: V0..V3 = K 8..15,  V4..V7 = K 24..31
__device__ __forceinline__ int a_frag_klocal(int half, int e) {
  int v = e >> 1, lo = e & 1;
  int base = (v < 4) ? ((half ? 8 : 0) + 2 * v)
                     : ((half ? 24 : 16) + 2 * (v - 4));
  return base + lo;
}

template <int N>
__device__ __forceinline__ void wait_asynccnt() {
#if defined(__AMDGCN__) && __has_builtin(__builtin_amdgcn_s_wait_asynccnt)
  __builtin_amdgcn_s_wait_asynccnt(N);
#else
  if constexpr (N == 0)      asm volatile("s_wait_asynccnt 0x0" ::: "memory");
  else if constexpr (N == 4) asm volatile("s_wait_asynccnt 0x4" ::: "memory");
  else                       asm volatile("s_wait_asynccnt 0x8" ::: "memory");
#endif
  asm volatile("" ::: "memory");
}

// Stage one K-step's A-fragment block ([MT][32 lanes][16 bf16]) into LDS.
template <int MT>
__device__ __forceinline__ void stage_weights(const unsigned short* __restrict__ packW,
                                              unsigned short* ldsBuf, int ks, int lane) {
#if HAS_ASYNC_LDS
  #pragma unroll
  for (int mt = 0; mt < MT; ++mt) {
    const unsigned short* s0 = packW + (((size_t)ks * MT + mt) * 32 + lane) * 16;
    unsigned short*       d0 = ldsBuf + (mt * 32 + lane) * 16;
    __builtin_amdgcn_global_load_async_to_lds_b128(
        (__attribute__((address_space(1))) v4i_t*)s0,
        (__attribute__((address_space(3))) v4i_t*)d0, 0, 0);
    __builtin_amdgcn_global_load_async_to_lds_b128(
        (__attribute__((address_space(1))) v4i_t*)(s0 + 8),
        (__attribute__((address_space(3))) v4i_t*)(d0 + 8), 0, 0);
  }
#else
  (void)packW; (void)ldsBuf; (void)ks; (void)lane;
#endif
}

// --------------------------- input normalize (with halo) -------------------
__global__ void convert_norm_pad_kernel(const float* __restrict__ x,
                                        unsigned short* __restrict__ y, int n) {
  constexpr int IN = 45, PIN = 47;
  int idx = blockIdx.x * blockDim.x + threadIdx.x;
  if (idx >= n) return;
  const int pin2 = PIN * PIN, pin3 = pin2 * PIN;
  int c = idx / pin3;
  int r = idx % pin3;
  int z  = r / pin2; r %= pin2;
  int yy = r / PIN;
  int xx = r % PIN;
  unsigned short v = 0;
  if (z >= 1 && z <= IN && yy >= 1 && yy <= IN && xx >= 1 && xx <= IN) {
    float f = x[(((size_t)c * IN + (z - 1)) * IN + (yy - 1)) * IN + (xx - 1)] * (1.0f / 255.0f);
    v = f32_to_bf16(f);
  }
  y[idx] = v;
}

__global__ void fill_zero_kernel(unsigned int* __restrict__ p, int nwords) {
  int i = blockIdx.x * blockDim.x + threadIdx.x;
  if (i < nwords) p[i] = 0u;
}

// --------------------------- im2col offset table (BYTE offsets) ------------
__global__ void build_ktab_kernel(int* __restrict__ tab, int KD, int KT,
                                  int Ktotal, int PIN, int n) {
  int k = blockIdx.x * blockDim.x + threadIdx.x;
  if (k >= n) return;
  int v = -1;                                   // sentinel: K padding -> zero
  if (k < Ktotal) {
    int ic = k / KT, tap = k % KT;
    int tz = tap / (KD * KD);
    int ty = (tap / KD) % KD;
    int tx = tap % KD;
    v = (((ic * PIN + tz) * PIN + ty) * PIN + tx) * 2;   // bytes
  }
  tab[k] = v;
}

// --------------------------- weight packing --------------------------------
// packed layout: [kstep][mtile][lane(32)][elem(16)] bf16
__global__ void pack_conv_w_kernel(const float* __restrict__ w,
                                   unsigned short* __restrict__ pw,
                                   int IC, int KT, int Ktotal, int mtiles, int total) {
  int idx = blockIdx.x * blockDim.x + threadIdx.x;
  if (idx >= total) return;
  int e     = idx & 15;
  int lane  = (idx >> 4) & 31;
  int mtile = (idx >> 9) % mtiles;
  int ks    = idx / (512 * mtiles);
  int half  = lane >> 4;
  int m     = (mtile << 4) + (lane & 15);
  int k     = ks * 32 + a_frag_klocal(half, e);
  float val = 0.0f;
  if (k < Ktotal) {
    int ic = k / KT, tap = k % KT;
    val = w[((size_t)m * IC + ic) * KT + tap];
  }
  pw[idx] = f32_to_bf16(val);
}

// --------------------------- conv as implicit GEMM -------------------------
// One wave per 16-position tile; computes MT 16-channel tiles (MT wmma per
// K-step sharing one B fragment). If POOL, positions are two full 2x2x2 pool
// windows; epilogue = bias+PReLU+max, bf16 out into next layer's halo layout.
template <int KD, bool POOL, int MT>
__global__ __launch_bounds__(32)
void conv3d_wmma_kernel(const unsigned short* __restrict__ in,
                        const unsigned short* __restrict__ packW,
                        const int* __restrict__ ktab,
                        const float* __restrict__ bias,
                        const float* __restrict__ slope,
                        unsigned short* __restrict__ actOut,
                        float* __restrict__ featOut,
                        int IC, int PIN, int POOLN, int outPad, int ksteps) {
  constexpr bool kAsync = (HAS_ASYNC_LDS != 0);
  const int lane  = threadIdx.x;
  const int half  = lane >> 4;
  const int nl    = lane & 15;
  const int s     = blockIdx.y;
  const int ntile = blockIdx.x;

  // conv output coordinate for this lane's B column (N = nl), in padded space
  int cz, cy, cx;
  const int PP = POOL ? POOLN * POOLN * POOLN : 0;
  if (POOL) {
    int q = nl >> 3, o = nl & 7;
    int p = ntile * 2 + q;
    if (p >= PP) p = PP - 1;                    // clamp: gather stays in-bounds
    int pz  = p / (POOLN * POOLN);
    int rem = p % (POOLN * POOLN);
    int py  = rem / POOLN, px = rem % POOLN;
    cz = 2 * pz + ((o >> 2) & 1);
    cy = 2 * py + ((o >> 1) & 1);
    cx = 2 * px + (o & 1);
  } else {
    int n = nl & 7;                             // 2x2x2 output volume
    cz = (n >> 2) & 1; cy = (n >> 1) & 1; cx = n & 1;
  }

  const int PIN2i = PIN * PIN;
  // wave-uniform sample base (SADDR) + 32-bit per-lane byte offset (voffset)
  const char* sbase = (const char*)in + (size_t)s * IC * PIN * PIN2i * 2;
  const int posB = ((cz * PIN + cy) * PIN + cx) * 2;   // bytes, per-lane

  v8f acc[MT];
  #pragma unroll
  for (int mt = 0; mt < MT; ++mt)
    #pragma unroll
    for (int i = 0; i < 8; ++i) acc[mt][i] = 0.0f;

  __shared__ __attribute__((aligned(32))) unsigned short wlds[2][MT * 512];

  if constexpr (kAsync) stage_weights<MT>(packW, &wlds[0][0], 0, lane);

  for (int ks = 0; ks < ksteps; ++ks) {
    // 16 im2col byte-offsets per lane in A-table K order (8h+0..7, 16+8h+0..7): four b128 loads
    const int kbase = ks * 32 + (half << 3);
    Int16 tt;
    const int4* tp = reinterpret_cast<const int4*>(ktab + kbase);
    #pragma unroll
    for (int j = 0; j < 4; ++j) tt.v[j] = tp[(j & 1) + ((j >> 1) << 2)];
    __builtin_prefetch(ktab + kbase + 32, 0, 1);   // global_prefetch_b8

    // B fragment: branch-free gather (sentinel -> 0), saddr + 32-bit voffset
    BF16x16 bfr;
    #pragma unroll
    for (int e = 0; e < 16; ++e) {
      int off  = tt.a[e];
      int moff = off < 0 ? 0 : off;
      unsigned short lv =
          *(const unsigned short*)(sbase + (unsigned)(posB + moff));
      bfr.u[e] = (off < 0) ? (unsigned short)0 : lv;
    }

    const int buf = ks & 1;
    if constexpr (kAsync) {
      if (ks + 1 < ksteps) {
        stage_weights<MT>(packW, &wlds[buf ^ 1][0], ks + 1, lane);
        wait_asynccnt<2 * MT>();                // current buffer complete
      } else {
        wait_asynccnt<0>();
      }
      #pragma unroll
      for (int mt = 0; mt < MT; ++mt) {
        v16bf a = *reinterpret_cast<const v16bf*>(&wlds[buf][(mt * 32 + lane) * 16]);
        acc[mt] = __builtin_amdgcn_wmma_f32_16x16x32_bf16(false, a, false, bfr.v,
                                                          (short)0, acc[mt], false, false);
      }
    } else {
      #pragma unroll
      for (int mt = 0; mt < MT; ++mt) {
        const size_t aoff = (((size_t)ks * MT + mt) * 32 + lane) * 16;
        v16bf a = *reinterpret_cast<const v16bf*>(packW + aoff);
        acc[mt] = __builtin_amdgcn_wmma_f32_16x16x32_bf16(false, a, false, bfr.v,
                                                          (short)0, acc[mt], false, false);
      }
    }
  }

  // stage C tiles in LDS: lane holds N=nl, rows M = r + 8*half
  __shared__ float cbuf[MT][16][16];
  #pragma unroll
  for (int mt = 0; mt < MT; ++mt)
    #pragma unroll
    for (int r = 0; r < 8; ++r) cbuf[mt][r + 8 * half][nl] = acc[mt][r];
  __syncthreads();

  const float sl = slope[0];
  if (POOL) {
    const int OC  = MT * 16;
    const int PO  = POOLN + 2 * outPad;
    const int PO2 = PO * PO, PO3 = PO2 * PO;
    int m = nl, q = half;
    int p = ntile * 2 + q;
    if (p < PP) {
      int pz  = p / (POOLN * POOLN);
      int rem = p % (POOLN * POOLN);
      int py  = rem / POOLN, px = rem % POOLN;
      size_t oidx = (size_t)(pz + outPad) * PO2 + (size_t)(py + outPad) * PO + (px + outPad);
      #pragma unroll
      for (int mt = 0; mt < MT; ++mt) {
        float bs   = bias[mt * 16 + m];
        float best = -3.0e38f;
        #pragma unroll
        for (int o = 0; o < 8; ++o) {
          float v = cbuf[mt][m][q * 8 + o] + bs;
          v = (v >= 0.0f) ? v : sl * v;         // PReLU then max (matches ref)
          best = fmaxf(best, v);
        }
        actOut[((size_t)s * OC + mt * 16 + m) * PO3 + oidx] = f32_to_bf16(best);
      }
    }
  } else {
    int m = nl;
    #pragma unroll
    for (int mt = 0; mt < MT; ++mt) {
      float bs = bias[mt * 16 + m];
      for (int n = half; n < 8; n += 2) {
        float v = cbuf[mt][m][n] + bs;
        v = (v >= 0.0f) ? v : sl * v;
        featOut[(size_t)s * 512 + (size_t)(mt * 16 + m) * 8 + n] = v;
      }
    }
  }
}

// --------------------------- CommNet FC layers -----------------------------
__global__ void mean_agents_kernel(const float* __restrict__ feat,
                                   float* __restrict__ m, int D) {
  int idx = blockIdx.x * blockDim.x + threadIdx.x;
  if (idx >= 16 * D) return;
  int d = idx % D, b = idx / D;
  float acc = 0.0f;
  #pragma unroll
  for (int a = 0; a < 5; ++a) acc += feat[((size_t)b * 5 + a) * D + d];
  m[idx] = acc * 0.2f;
}

__global__ void comm_fc_kernel(const float* __restrict__ feat,
                               const float* __restrict__ mean,
                               const float* __restrict__ w,
                               const float* __restrict__ b,
                               const float* __restrict__ slope,
                               float* __restrict__ out, int D, int O) {
  int idx = blockIdx.x * blockDim.x + threadIdx.x;
  if (idx >= 16 * 5 * O) return;
  int o  = idx % O;
  int a  = (idx / O) % 5;
  int bb = idx / (O * 5);
  const float* wr = w + ((size_t)a * O + o) * 2 * D;
  const float* fr = feat + ((size_t)bb * 5 + a) * D;
  const float* mr = mean + (size_t)bb * D;
  float acc = b[a * O + o];
  for (int i = 0; i < D; ++i) acc = fmaf(fr[i], wr[i], acc);
  for (int i = 0; i < D; ++i) acc = fmaf(mr[i], wr[D + i], acc);
  if (slope) { float sl = slope[0]; acc = (acc >= 0.0f) ? acc : sl * acc; }
  out[idx] = acc;
}

// --------------------------- launcher --------------------------------------
extern "C" void kernel_launch(void* const* d_in, const int* in_sizes, int n_in,
                              void* d_out, int out_size, void* d_ws, size_t ws_size,
                              hipStream_t stream) {
  const float* x    = (const float*)d_in[0];
  const float* w0   = (const float*)d_in[1];
  const float* b0   = (const float*)d_in[2];
  const float* p0   = (const float*)d_in[3];
  const float* w1   = (const float*)d_in[4];
  const float* b1   = (const float*)d_in[5];
  const float* p1   = (const float*)d_in[6];
  const float* w2   = (const float*)d_in[7];
  const float* b2   = (const float*)d_in[8];
  const float* p2   = (const float*)d_in[9];
  const float* w3   = (const float*)d_in[10];
  const float* b3   = (const float*)d_in[11];
  const float* p3   = (const float*)d_in[12];
  const float* fc1w = (const float*)d_in[13];
  const float* fc1b = (const float*)d_in[14];
  const float* p4   = (const float*)d_in[15];
  const float* fc2w = (const float*)d_in[16];
  const float* fc2b = (const float*)d_in[17];
  const float* p5   = (const float*)d_in[18];
  const float* fc3w = (const float*)d_in[19];
  const float* fc3b = (const float*)d_in[20];
  float* out = (float*)d_out;

  char*  ws  = (char*)d_ws;
  size_t off = 0;
  auto alloc = [&](size_t bytes) -> void* {
    void* p = ws + off;
    off = (off + bytes + 255) & ~(size_t)255;
    return p;
  };
  // halo-padded activations (bf16 as u16)
  const size_t nXbf  = 320ull * 47 * 47 * 47;        // 80*4  @ 47^3
  const size_t nAct0 = 2560ull * 23 * 23 * 23;       // 80*32 @ 23^3
  const size_t nAct1 = 2560ull * 11 * 11 * 11;       // 80*32 @ 11^3
  const size_t nAct2 = 5120ull * 4 * 4 * 4;          // 80*64 @ 4^3 (no halo)
  unsigned short* xbf  = (unsigned short*)alloc(nXbf * 2);
  unsigned short* act0 = (unsigned short*)alloc(nAct0 * 2);
  unsigned short* act1 = (unsigned short*)alloc(nAct1 * 2);
  unsigned short* act2 = (unsigned short*)alloc(nAct2 * 2);
  float* feat  = (float*)alloc(80ull * 512 * 4);
  float* mean1 = (float*)alloc(16ull * 512 * 4);
  float* fc1o  = (float*)alloc(16ull * 5 * 256 * 4);
  float* mean2 = (float*)alloc(16ull * 256 * 4);
  float* fc2o  = (float*)alloc(16ull * 5 * 128 * 4);
  float* mean3 = (float*)alloc(16ull * 128 * 4);
  unsigned short* pw0 = (unsigned short*)alloc( 16ull * 2 * 512 * 2);
  unsigned short* pw1 = (unsigned short*)alloc(125ull * 2 * 512 * 2);
  unsigned short* pw2 = (unsigned short*)alloc( 64ull * 4 * 512 * 2);
  unsigned short* pw3 = (unsigned short*)alloc( 54ull * 4 * 512 * 2);
  int* ktab0 = (int*)alloc( 512ull * 4);
  int* ktab1 = (int*)alloc(4000ull * 4);
  int* ktab2 = (int*)alloc(2048ull * 4);
  int* ktab3 = (int*)alloc(1728ull * 4);
  (void)ws_size; (void)in_sizes; (void)n_in; (void)out_size;

  // 1) x/255 -> bf16 with zero halo; zero-fill downstream halo buffers
  {
    int n = (int)nXbf;
    convert_norm_pad_kernel<<<(n + 255) / 256, 256, 0, stream>>>(x, xbf, n);
    int f0 = (int)(nAct0 / 2), f1 = (int)(nAct1 / 2);
    fill_zero_kernel<<<(f0 + 255) / 256, 256, 0, stream>>>((unsigned int*)act0, f0);
    fill_zero_kernel<<<(f1 + 255) / 256, 256, 0, stream>>>((unsigned int*)act1, f1);
  }
  // 2) im2col offset tables + packed A-fragment weights
  {
    build_ktab_kernel<<<2, 256, 0, stream>>>(ktab0, 5, 125,  500, 47,  512);
    build_ktab_kernel<<<16, 256, 0, stream>>>(ktab1, 5, 125, 4000, 23, 4000);
    build_ktab_kernel<<<8, 256, 0, stream>>>(ktab2, 4,  64, 2048, 11, 2048);
    build_ktab_kernel<<<7, 256, 0, stream>>>(ktab3, 3,  27, 1728,  4, 1728);
    int t0 = 16 * 2 * 512, t1 = 125 * 2 * 512, t2 = 64 * 4 * 512, t3 = 54 * 4 * 512;
    pack_conv_w_kernel<<<(t0 + 255) / 256, 256, 0, stream>>>(w0, pw0,  4, 125,  500, 2, t0);
    pack_conv_w_kernel<<<(t1 + 255) / 256, 256, 0, stream>>>(w1, pw1, 32, 125, 4000, 2, t1);
    pack_conv_w_kernel<<<(t2 + 255) / 256, 256, 0, stream>>>(w2, pw2, 32,  64, 2048, 4, t2);
    pack_conv_w_kernel<<<(t3 + 255) / 256, 256, 0, stream>>>(w3, pw3, 64,  27, 1728, 4, t3);
  }
  // 3) conv stack (WMMA implicit GEMM, fused PReLU+pool, halo-layout outputs)
  {
    dim3 g0((21 * 21 * 21 + 1) / 2, 80);
    conv3d_wmma_kernel<5, true, 2><<<g0, 32, 0, stream>>>(
        xbf, pw0, ktab0, b0, p0, act0, nullptr, 4, 47, 21, 1, 16);
    dim3 g1((9 * 9 * 9 + 1) / 2, 80);
    conv3d_wmma_kernel<5, true, 2><<<g1, 32, 0, stream>>>(
        act0, pw1, ktab1, b1, p1, act1, nullptr, 32, 23, 9, 1, 125);
    dim3 g2((4 * 4 * 4) / 2, 80);
    conv3d_wmma_kernel<4, true, 4><<<g2, 32, 0, stream>>>(
        act1, pw2, ktab2, b2, p2, act2, nullptr, 32, 11, 4, 0, 64);
    dim3 g3(1, 80);
    conv3d_wmma_kernel<3, false, 4><<<g3, 32, 0, stream>>>(
        act2, pw3, ktab3, b3, p3, nullptr, feat, 64, 4, 0, 0, 54);
  }
  // 4) CommNet FC stack
  {
    int n;
    n = 16 * 512;
    mean_agents_kernel<<<(n + 255) / 256, 256, 0, stream>>>(feat, mean1, 512);
    n = 16 * 5 * 256;
    comm_fc_kernel<<<(n + 255) / 256, 256, 0, stream>>>(feat, mean1, fc1w, fc1b, p4, fc1o, 512, 256);
    n = 16 * 256;
    mean_agents_kernel<<<(n + 255) / 256, 256, 0, stream>>>(fc1o, mean2, 256);
    n = 16 * 5 * 128;
    comm_fc_kernel<<<(n + 255) / 256, 256, 0, stream>>>(fc1o, mean2, fc2w, fc2b, p5, fc2o, 256, 128);
    n = 16 * 128;
    mean_agents_kernel<<<(n + 255) / 256, 256, 0, stream>>>(fc2o, mean3, 128);
    n = 16 * 5 * 45;
    comm_fc_kernel<<<(n + 255) / 256, 256, 0, stream>>>(fc2o, mean3, fc3w, fc3b, nullptr, out, 128, 45);
  }
}
